// FlashAttentionModel_38543036514572
// MI455X (gfx1250) — matrix operand order repaired and checked by reading, not compiled, not run on hardware
//
#include <hip/hip_runtime.h>

// ---------------- CDNA5 WMMA types & helpers ----------------
typedef __attribute__((ext_vector_type(16))) __bf16 v16bf;
typedef __attribute__((ext_vector_type(8)))  float  v8f;

union BF16Frag { v16bf v; unsigned short u[16]; };
union F32Acc   { v8f   v; float          f[8];  };

__device__ __forceinline__ unsigned short f2bf(float f) {
  unsigned int u = __float_as_uint(f);
  u += 0x7FFFu + ((u >> 16) & 1u);          // round-to-nearest-even
  return (unsigned short)(u >> 16);
}

// ---------------- constants ----------------
// B=1, S=2048, D=512, H=8, HD=64, F=2048, E=8, K=2, L=4, V=50257
#define CS 2048
#define CD 512
#define CH 8
#define CHD 64
#define CF 2048
#define CE 8
#define CL 4
#define CV 50257

// =====================================================================
// bf16 WMMA GEMM:  C[M,N] = act( A[M,K](bf16) * B[K,N](f32->bf16) )
// TRANSB: B stored [N,K].  SILU epilogue option.  Optional bf16 mirror
// (Cb) or fused scaled-accumulate (accScale: C += accScale[m*E] * v).
// Block: 256 thr = 8 waves (4Mx2N); tile 64(M) x 128(N); K step 32,
// double-buffered (2x32) software pipeline:
//   - A tile streamed by GLOBAL_LOAD_ASYNC_TO_LDS_B128 (ASYNCcnt)
//   - B tile: global loads issued -> WMMA on current buffer -> convert
//     + ds_store into next buffer (load-wait hidden behind WMMAs)
// Requires: M % 64 == 0, Kd % 64 == 0 (true for all calls here).
// =====================================================================
template<int TRANSB, int SILU>
__global__ __launch_bounds__(256)
void k_gemm(const unsigned short* __restrict__ A, const float* __restrict__ B,
            float* __restrict__ C, unsigned short* __restrict__ Cb,
            const float* __restrict__ accScale, int M, int N, int Kd) {
  __shared__ alignas(16) unsigned short As[2][64][40];   // [buf][m][k]
  __shared__ alignas(16) unsigned short Bs[2][128][40];  // [buf][n][k]

  const int tid  = threadIdx.x;
  const int lane = tid & 31;
  const int wave = tid >> 5;
  const int half = lane >> 4;               // 0/1
  const int mr   = lane & 15;
  const int wm   = wave & 3;                // 0..3 -> M subtile
  const int wn   = wave >> 2;               // 0..1 -> N half
  const int m0   = blockIdx.y * 64;
  const int n0   = blockIdx.x * 128;
  const bool full = (n0 + 128 <= N);        // uniform per block

  // staging coordinates
  const int ar  = tid >> 2;                 // A row 0..63
  const int as  = (tid & 3) * 8;            // A k-offset (8 u16 = 16B)
  const int bnI = TRANSB ? (tid >> 1) : (tid & 127);             // B n
  const int bkI = TRANSB ? ((tid & 1) * 16) : ((tid >> 7) * 16); // B k base

  F32Acc acc[4];
#pragma unroll
  for (int c = 0; c < 4; ++c)
#pragma unroll
    for (int i = 0; i < 8; ++i) acc[c].f[i] = 0.f;

  // --- pipeline helpers ---
  auto stageA = [&](unsigned short (*Asb)[40], int k0) {
    // async DMA: 16B per lane, LDS address = low 32 bits of generic ptr
    unsigned int ldsOff = (unsigned int)(size_t)&Asb[ar][as];
    const unsigned short* src = &A[(size_t)(m0 + ar) * Kd + k0 + as];
    asm volatile("global_load_async_to_lds_b128 %0, %1, off"
                 :: "v"(ldsOff), "v"(src) : "memory");
  };
  auto loadB = [&](float* rB, int k0) {
    const int gn = n0 + bnI;
    if (TRANSB) {
      if (full || gn < N) {
        const float* src = &B[(size_t)gn * Kd + k0 + bkI];
#pragma unroll
        for (int i = 0; i < 16; ++i) rB[i] = src[i];
      } else {
#pragma unroll
        for (int i = 0; i < 16; ++i) rB[i] = 0.f;
      }
    } else {
      if (full || gn < N) {
#pragma unroll
        for (int i = 0; i < 16; ++i) rB[i] = B[(size_t)(k0 + bkI + i) * N + gn];
      } else {
#pragma unroll
        for (int i = 0; i < 16; ++i) rB[i] = 0.f;
      }
    }
  };
  auto storeB = [&](unsigned short (*Bsb)[40], const float* rB) {
    alignas(16) unsigned short tmp[16];
#pragma unroll
    for (int i = 0; i < 16; ++i) tmp[i] = f2bf(rB[i]);
    ((uint4*)&Bsb[bnI][bkI])[0] = ((const uint4*)tmp)[0];
    ((uint4*)&Bsb[bnI][bkI])[1] = ((const uint4*)tmp)[1];
  };
  auto compute = [&](const unsigned short (*Asb)[40],
                     const unsigned short (*Bsb)[40]) {
    BF16Frag a;
#pragma unroll
    for (int e = 0; e < 16; ++e) {
      int kk = half * 16 + e;
      a.u[e] = Asb[wm * 16 + mr][kk];
    }
#pragma unroll
    for (int c = 0; c < 4; ++c) {
      BF16Frag b;
#pragma unroll
      for (int e = 0; e < 16; ++e)
        b.u[e] = Bsb[wn * 64 + c * 16 + mr][half * 16 + e];
      acc[c].v = __builtin_amdgcn_wmma_f32_16x16x32_bf16(
          false, a.v, false, b.v, (short)0, acc[c].v, false, false);
    }
  };
  auto waitAsync = [] { asm volatile("s_wait_asynccnt 0" ::: "memory"); };

  // --- prologue: stage K-step 0 into buffer 0 ---
  float rB[16];
  loadB(rB, 0);
  stageA(As[0], 0);
  storeB(Bs[0], rB);
  waitAsync();
  __syncthreads();

  // --- double-buffered main loop (Kd % 64 == 0) ---
  for (int k0 = 0; k0 < Kd; k0 += 64) {
    loadB(rB, k0 + 32);
    stageA(As[1], k0 + 32);
    compute(As[0], Bs[0]);
    storeB(Bs[1], rB);
    waitAsync();
    __syncthreads();

    const bool more = (k0 + 64 < Kd);
    if (more) { loadB(rB, k0 + 64); stageA(As[0], k0 + 64); }
    compute(As[1], Bs[1]);
    if (more) storeB(Bs[0], rB);
    waitAsync();
    __syncthreads();
  }

  // ---- epilogue ----
  const int mB = m0 + wm * 16 + half * 8;
#pragma unroll
  for (int c = 0; c < 4; ++c) {
    const int n = n0 + wn * 64 + c * 16 + mr;
    if (n < N) {
#pragma unroll
      for (int r = 0; r < 8; ++r) {
        float v = acc[c].f[r];
        if (SILU) v = v * (1.f / (1.f + __expf(-v)));
        size_t o = (size_t)(mB + r) * N + n;
        if (accScale) {
          C[o] += accScale[(size_t)(mB + r) * CE] * v;
        } else {
          C[o] = v;
          if (Cb) Cb[o] = f2bf(v);
        }
      }
    }
  }
}

// =====================================================================
// Flash attention, causal, RoPE'd bf16 q/k/v.  One wave per 16-query
// block per head; 32-key blocks with online softmax; WMMA for QK^T & PV.
// =====================================================================
__global__ __launch_bounds__(32)
void k_attn(const unsigned short* __restrict__ qbp, const unsigned short* __restrict__ kbp,
            const unsigned short* __restrict__ vbp, float* __restrict__ attn,
            unsigned short* __restrict__ attnb) {
  __shared__ alignas(16) unsigned short Ks[32][64];   // [key][d]
  __shared__ unsigned short Vt[64][33];               // [d][key] transposed
  __shared__ float          Sc[16][32];
  __shared__ unsigned short Pb[16][32];
  __shared__ float          sScale[16];
  __shared__ float          sL[16];

  const int lane = threadIdx.x;
  const int half = lane >> 4, mr = lane & 15;
  const int q0   = blockIdx.x * 16;
  const int hb   = blockIdx.y * CHD;

  BF16Frag aq0, aq1;
#pragma unroll
  for (int e = 0; e < 16; ++e) {
    int kk = half * 16 + e;
    aq0.u[e] = qbp[(size_t)(q0 + mr) * CD + hb + kk];
    aq1.u[e] = qbp[(size_t)(q0 + mr) * CD + hb + 32 + kk];
  }

  F32Acc o[4];
#pragma unroll
  for (int t = 0; t < 4; ++t)
#pragma unroll
    for (int i = 0; i < 8; ++i) o[t].f[i] = 0.f;
  float mI = -1e30f, lI = 0.f;              // lanes 0..15 own row = lane

  for (int kb0 = 0; kb0 <= q0 + 15; kb0 += 32) {
    const int keyRow = kb0 + lane;
    const unsigned short* kr = &kbp[(size_t)keyRow * CD + hb];
    const unsigned short* vr = &vbp[(size_t)keyRow * CD + hb];
#pragma unroll
    for (int s = 0; s < 8; ++s)
      *(uint4*)&Ks[lane][s * 8] = *(const uint4*)(kr + s * 8);
#pragma unroll
    for (int d = 0; d < 64; ++d) Vt[d][lane] = vr[d];
    __syncthreads();

    // scores = Q K^T : 16x32 as two 16x16 tiles, Kdim 64
#pragma unroll
    for (int c = 0; c < 2; ++c) {
      F32Acc s;
#pragma unroll
      for (int i = 0; i < 8; ++i) s.f[i] = 0.f;
      BF16Frag bk;
#pragma unroll
      for (int e = 0; e < 16; ++e) bk.u[e] = Ks[c * 16 + mr][half * 16 + e];
      s.v = __builtin_amdgcn_wmma_f32_16x16x32_bf16(false, aq0.v, false, bk.v,
                                                    (short)0, s.v, false, false);
#pragma unroll
      for (int e = 0; e < 16; ++e) bk.u[e] = Ks[c * 16 + mr][32 + half * 16 + e];
      s.v = __builtin_amdgcn_wmma_f32_16x16x32_bf16(false, aq1.v, false, bk.v,
                                                    (short)0, s.v, false, false);
#pragma unroll
      for (int r = 0; r < 8; ++r)
        Sc[half * 8 + r][c * 16 + mr] = s.f[r] * 0.125f;    // 1/sqrt(64)
    }
    __syncthreads();

    if (lane < 16) {                        // online softmax, one row/lane
      const int qi = q0 + lane;
      float rowmax = mI;
      for (int j = 0; j < 32; ++j)
        if (kb0 + j <= qi) rowmax = fmaxf(rowmax, Sc[lane][j]);
      float sc = __expf(mI - rowmax);
      float ls = 0.f;
      for (int j = 0; j < 32; ++j) {
        float p = 0.f;
        if (kb0 + j <= qi) p = __expf(Sc[lane][j] - rowmax);
        Pb[lane][j] = f2bf(p);
        ls += p;
      }
      mI = rowmax;
      lI = lI * sc + ls;
      sScale[lane] = sc;
    }
    __syncthreads();

    BF16Frag ap;
#pragma unroll
    for (int e = 0; e < 16; ++e) {
      int kk = half * 16 + e;
      ap.u[e] = Pb[mr][kk];
    }
#pragma unroll
    for (int t = 0; t < 4; ++t) {           // O = diag(scale)*O + P*V
#pragma unroll
      for (int r = 0; r < 8; ++r) o[t].f[r] *= sScale[half * 8 + r];
      BF16Frag bv;
#pragma unroll
      for (int e = 0; e < 16; ++e) bv.u[e] = Vt[t * 16 + mr][half * 16 + e];
      o[t].v = __builtin_amdgcn_wmma_f32_16x16x32_bf16(false, ap.v, false, bv.v,
                                                       (short)0, o[t].v, false, false);
    }
    __syncthreads();
  }

  if (lane < 16) sL[lane] = lI;
  __syncthreads();
#pragma unroll
  for (int t = 0; t < 4; ++t)
#pragma unroll
    for (int r = 0; r < 8; ++r) {
      int m = half * 8 + r;
      float v = o[t].f[r] / sL[m];
      size_t off = (size_t)(q0 + m) * CD + hb + t * 16 + mr;
      attn[off]  = v;
      attnb[off] = f2bf(v);
    }
}

// =====================================================================
// elementwise / small kernels
// =====================================================================
__global__ __launch_bounds__(256)
void k_embed(const int* __restrict__ tokens, const float* __restrict__ emb,
             float* __restrict__ x, unsigned short* __restrict__ xb) {
  int idx = blockIdx.x * 256 + threadIdx.x;
  int t = idx >> 9, d = idx & 511;
  float v = emb[(size_t)tokens[t] * CD + d] * 22.62741699796952f;  // sqrt(512)
  x[idx]  = v;
  xb[idx] = f2bf(v);
}

__global__ __launch_bounds__(256)
void k_rope(const float* __restrict__ q, const float* __restrict__ k,
            const float* __restrict__ v, unsigned short* __restrict__ qb,
            unsigned short* __restrict__ kb, unsigned short* __restrict__ vb) {
  int idx = blockIdx.x * 256 + threadIdx.x;
  int s = idx >> 9;
  int c = idx & 511;
  int d = c & 63;
  int j = d & 31;
  float inv = __powf(10000.f, -(float)j * (1.f / 32.f));
  float ang = (float)s * inv;
  float co = __cosf(ang), si = __sinf(ang);
  float qv = q[idx], kv = k[idx], qo, ko;
  if (d < 32) { qo = qv * co - q[idx + 32] * si; ko = kv * co - k[idx + 32] * si; }
  else        { qo = qv * co + q[idx - 32] * si; ko = kv * co + k[idx - 32] * si; }
  qb[idx] = f2bf(qo); kb[idx] = f2bf(ko); vb[idx] = f2bf(v[idx]);
}

__global__ __launch_bounds__(256)
void k_ln(float* __restrict__ x, const float* __restrict__ a,
          const float* __restrict__ g, const float* __restrict__ b,
          unsigned short* __restrict__ xb) {
  __shared__ float red[256];
  const int t = blockIdx.x, tid = threadIdx.x;
  float v0 = x[(size_t)t * CD + tid]       + a[(size_t)t * CD + tid];
  float v1 = x[(size_t)t * CD + 256 + tid] + a[(size_t)t * CD + 256 + tid];
  red[tid] = v0 + v1;
  __syncthreads();
  for (int s = 128; s > 0; s >>= 1) { if (tid < s) red[tid] += red[tid + s]; __syncthreads(); }
  float mu = red[0] * (1.f / 512.f);
  __syncthreads();
  float d0 = v0 - mu, d1 = v1 - mu;
  red[tid] = d0 * d0 + d1 * d1;
  __syncthreads();
  for (int s = 128; s > 0; s >>= 1) { if (tid < s) red[tid] += red[tid + s]; __syncthreads(); }
  float rinv = rsqrtf(red[0] * (1.f / 512.f) + 1e-5f);
  float o0 = d0 * rinv * g[tid]       + b[tid];
  float o1 = d1 * rinv * g[256 + tid] + b[256 + tid];
  x[(size_t)t * CD + tid] = o0;        xb[(size_t)t * CD + tid] = f2bf(o0);
  x[(size_t)t * CD + 256 + tid] = o1;  xb[(size_t)t * CD + 256 + tid] = f2bf(o1);
}

__global__ __launch_bounds__(256)
void k_rms(const float* __restrict__ x, const float* __restrict__ g,
           unsigned short* __restrict__ xb) {
  __shared__ float red[256];
  const int t = blockIdx.x, tid = threadIdx.x;
  float v0 = x[(size_t)t * CD + tid];
  float v1 = x[(size_t)t * CD + 256 + tid];
  red[tid] = v0 * v0 + v1 * v1;
  __syncthreads();
  for (int s = 128; s > 0; s >>= 1) { if (tid < s) red[tid] += red[tid + s]; __syncthreads(); }
  float rinv = rsqrtf(red[0] * (1.f / 512.f) + 1e-5f);
  xb[(size_t)t * CD + tid]       = f2bf(v0 * rinv * g[tid]);
  xb[(size_t)t * CD + 256 + tid] = f2bf(v1 * rinv * g[256 + tid]);
}

__global__ __launch_bounds__(32)
void k_router(const float* __restrict__ x, const float* __restrict__ rw,
              float* __restrict__ w, float* __restrict__ probs,
              float* __restrict__ cnt, float* __restrict__ psum) {
  __shared__ float lg[CE];
  const int t = blockIdx.x, lane = threadIdx.x;
  if (lane < CE) {
    float s = 0.f;
    for (int d = 0; d < CD; ++d) s += x[(size_t)t * CD + d] * rw[(size_t)d * CE + lane];
    lg[lane] = s;
  }
  __syncthreads();
  if (lane == 0) {
    float mx = lg[0];
    for (int e = 1; e < CE; ++e) mx = fmaxf(mx, lg[e]);
    float p[CE], sum = 0.f;
    for (int e = 0; e < CE; ++e) { p[e] = __expf(lg[e] - mx); sum += p[e]; }
    float rs = 1.f / sum;
    for (int e = 0; e < CE; ++e) {
      p[e] *= rs;
      probs[(size_t)t * CE + e] = p[e];
      atomicAdd(&psum[e], p[e]);
      w[(size_t)t * CE + e] = 0.f;
    }
    int i1 = 0;
    for (int e = 1; e < CE; ++e) if (p[e] > p[i1]) i1 = e;
    int i2 = (i1 == 0) ? 1 : 0;
    for (int e = 0; e < CE; ++e) if (e != i1 && p[e] > p[i2]) i2 = e;
    float tot = p[i1] + p[i2];
    w[(size_t)t * CE + i1] = p[i1] / tot;
    w[(size_t)t * CE + i2] = p[i2] / tot;
    atomicAdd(&cnt[i1], 1.f);
    atomicAdd(&cnt[i2], 1.f);
  }
}

__global__ void k_aux(const float* __restrict__ cnt, const float* __restrict__ psum,
                      float* __restrict__ outAux) {
  if (threadIdx.x == 0 && blockIdx.x == 0) {
    float total = 0.f;
    for (int l = 0; l < CL; ++l) {
      float s = 0.f;
      for (int e = 0; e < CE; ++e)
        s += (cnt[l * CE + e] * (1.f / (float)CS)) * (psum[l * CE + e] * (1.f / (float)CS));
      total += (float)CE * s;
    }
    *outAux = total;
  }
}

// =====================================================================
// host orchestration
// =====================================================================
extern "C" void kernel_launch(void* const* d_in, const int* in_sizes, int n_in,
                              void* d_out, int out_size, void* d_ws, size_t ws_size,
                              hipStream_t stream) {
  (void)in_sizes; (void)n_in; (void)out_size; (void)ws_size;
  constexpr int S = CS, D = CD, H = CH, F = CF, E = CE, L = CL, V = CV;

  const int*   tokens = (const int*)  d_in[0];
  const float* emb    = (const float*)d_in[1];
  const float* wq     = (const float*)d_in[2];
  const float* wk     = (const float*)d_in[3];
  const float* wv     = (const float*)d_in[4];
  const float* wo     = (const float*)d_in[5];
  const float* ln1g   = (const float*)d_in[6];
  const float* ln1b   = (const float*)d_in[7];
  const float* ln2g   = (const float*)d_in[8];
  const float* ln2b   = (const float*)d_in[9];
  const float* rwAll  = (const float*)d_in[10];
  const float* w1All  = (const float*)d_in[11];
  const float* w2All  = (const float*)d_in[12];
  const float* fg     = (const float*)d_in[13];
  float* out = (float*)d_out;

  char* p = (char*)d_ws;
  auto alloc = [&](size_t bytes) -> void* {
    void* r = (void*)p;
    p += (bytes + 255) & ~(size_t)255;
    return r;
  };
  float* x      = (float*)alloc((size_t)S * D * 4);
  float* qf     = (float*)alloc((size_t)S * D * 4);
  float* kf     = (float*)alloc((size_t)S * D * 4);
  float* vf     = (float*)alloc((size_t)S * D * 4);
  float* attn   = (float*)alloc((size_t)S * D * 4);
  float* proj   = (float*)alloc((size_t)S * D * 4);
  float* moey   = (float*)alloc((size_t)S * D * 4);
  float* h      = (float*)alloc((size_t)S * F * 4);
  unsigned short* xb    = (unsigned short*)alloc((size_t)S * D * 2);
  unsigned short* qb    = (unsigned short*)alloc((size_t)S * D * 2);
  unsigned short* kb    = (unsigned short*)alloc((size_t)S * D * 2);
  unsigned short* vb    = (unsigned short*)alloc((size_t)S * D * 2);
  unsigned short* attnb = (unsigned short*)alloc((size_t)S * D * 2);
  unsigned short* hb    = (unsigned short*)alloc((size_t)S * F * 2);
  float* wbuf  = (float*)alloc((size_t)S * E * 4);
  float* probs = (float*)alloc((size_t)S * E * 4);
  float* stats = (float*)alloc((size_t)2 * L * E * 4);
  float* cnt  = stats;
  float* psum = stats + (size_t)L * E;

  auto run = [&](auto kern, const unsigned short* A, const float* B, float* C,
                 unsigned short* Cb, const float* accS, int M, int N, int Kd) {
    dim3 g((N + 127) / 128, M / 64), b(256);
    kern<<<g, b, 0, stream>>>(A, B, C, Cb, accS, M, N, Kd);
  };

  hipMemsetAsync(stats, 0, (size_t)2 * L * E * 4, stream);
  k_embed<<<(S * D) / 256, 256, 0, stream>>>(tokens, emb, x, xb);

  for (int l = 0; l < L; ++l) {
    const float* WQ = wq + (size_t)l * D * D;
    const float* WK = wk + (size_t)l * D * D;
    const float* WV = wv + (size_t)l * D * D;
    const float* WO = wo + (size_t)l * D * D;

    run(k_gemm<0,0>, xb, WQ, qf, nullptr, nullptr, S, D, D);
    run(k_gemm<0,0>, xb, WK, kf, nullptr, nullptr, S, D, D);
    run(k_gemm<0,0>, xb, WV, vf, nullptr, nullptr, S, D, D);
    k_rope<<<(S * D) / 256, 256, 0, stream>>>(qf, kf, vf, qb, kb, vb);
    k_attn<<<dim3(S / 16, H), 32, 0, stream>>>(qb, kb, vb, attn, attnb);
    run(k_gemm<0,0>, attnb, WO, proj, nullptr, nullptr, S, D, D);
    k_ln<<<S, 256, 0, stream>>>(x, proj, ln1g + (size_t)l * D, ln1b + (size_t)l * D, xb);

    k_router<<<S, 32, 0, stream>>>(x, rwAll + (size_t)l * D * E, wbuf, probs,
                                   cnt + (size_t)l * E, psum + (size_t)l * E);
    hipMemsetAsync(moey, 0, (size_t)S * D * 4, stream);
    for (int e = 0; e < E; ++e) {
      const float* W1 = w1All + ((size_t)l * E + e) * D * F;
      const float* W2 = w2All + ((size_t)l * E + e) * F * D;
      run(k_gemm<0,1>, xb, W1, h, hb, nullptr, S, F, D);          // SiLU + bf16 mirror
      run(k_gemm<0,0>, hb, W2, moey, nullptr, wbuf + e, S, D, F); // fused w[t,e]*h2 accum
    }
    k_ln<<<S, 256, 0, stream>>>(x, moey, ln2g + (size_t)l * D, ln2b + (size_t)l * D, xb);
  }

  k_rms<<<S, 256, 0, stream>>>(x, fg, xb);
  run(k_gemm<1,0>, xb, emb, out, nullptr, nullptr, S, V, D);      // tied lm head
  k_aux<<<1, 1, 0, stream>>>(cnt, psum, out + (size_t)S * V);
}
